// BiDAF_44968307589585
// MI455X (gfx1250) — matrix operand order repaired and checked by reading, not compiled, not run on hardware
//
#include <hip/hip_runtime.h>
#include <hip/hip_bf16.h>

// ---------------------------------------------------------------------------
// BiDAF forward for MI455X (gfx1250, wave32, WMMA bf16 + async-LDS path)
// ---------------------------------------------------------------------------

typedef __attribute__((ext_vector_type(16))) __bf16 v16bf;
typedef __attribute__((ext_vector_type(8)))  float  v8f;
typedef unsigned short u16b;

constexpr int Hh  = 256;    // hidden
constexpr int Dm  = 512;    // 2H
constexpr int DD  = 2048;   // 8H
constexpr int Bb  = 32;     // batch
constexpr int CLl = 400;    // context length
constexpr int QLl = 64;     // query length
constexpr int WLl = 16;     // word length (chars)
constexpr int CDd = 16;     // char emb dim
constexpr int CCc = 212;    // conv channels
constexpr int CWw = 5;      // conv width
constexpr int WDd = 300;    // word emb dim

__device__ __forceinline__ u16b f2bf(float f) {
  unsigned u = __float_as_uint(f);
  unsigned r = u + 0x7FFFu + ((u >> 16) & 1u);
  return (u16b)(r >> 16);
}
__device__ __forceinline__ float sigm(float x) { return 1.0f / (1.0f + __expf(-x)); }

struct Frag { union { v16bf v; unsigned u[8]; u16b s[16]; }; };

// CDNA5 async global->LDS copy (16B per lane), tracked by ASYNCcnt.
__device__ __forceinline__ void async_ld_b128(unsigned ldsoff, const u16b* g) {
  asm volatile("global_load_async_to_lds_b128 %0, %1, off"
               :: "v"(ldsoff), "v"((unsigned long long)(uintptr_t)g)
               : "memory");
}
__device__ __forceinline__ void wait_async3() {
  asm volatile("s_wait_asynccnt 0x3" ::: "memory");
}
__device__ __forceinline__ void wait_async0() {
  asm volatile("s_wait_asynccnt 0x0" ::: "memory");
}

// ---------------------------------------------------------------------------
// elementwise / conversion kernels
// ---------------------------------------------------------------------------
__global__ void k_f32_to_bf16(const float* __restrict__ a, u16b* __restrict__ b, int n) {
  int i = blockIdx.x * 256 + threadIdx.x;
  if (i < n) b[i] = f2bf(a[i]);
}

__global__ void k_scale_bf16(const float* __restrict__ a, const float* __restrict__ w,
                             u16b* __restrict__ b, int n, int mask) {
  int i = blockIdx.x * 256 + threadIdx.x;
  if (i < n) b[i] = f2bf(a[i] * w[i & mask]);
}

// char conv + maxpool: one block per (b,s); 256 threads
__global__ void k_char_conv(const int* __restrict__ idx, const float* __restrict__ emb,
                            const float* __restrict__ cw, const float* __restrict__ cb,
                            float* __restrict__ feat) {
  __shared__ float e[WLl][CDd];
  int bs = blockIdx.x;
  int t = threadIdx.x;
  {
    int w = t >> 4, d = t & 15;          // 256 threads == 16x16
    int ci = idx[bs * WLl + w];
    e[w][d] = emb[ci * CDd + d];
  }
  __syncthreads();
  int cc = t;
  if (cc < CCc) {
    float bias = cb[cc];
    float mx = -3.4e38f;
    for (int p = 0; p < WLl - CWw + 1; p++) {
      float acc = bias;
#pragma unroll
      for (int w = 0; w < CWw; w++)
#pragma unroll
        for (int d = 0; d < CDd; d++)
          acc += e[p + w][d] * cw[cc * (CDd * CWw) + d * CWw + w];
      mx = fmaxf(mx, acc);
    }
    feat[bs * CCc + cc] = mx;
  }
}

__global__ void k_concat(const float* __restrict__ feat, const int* __restrict__ word,
                         const float* __restrict__ wemb, float* __restrict__ x,
                         u16b* __restrict__ xb, int BS) {
  int i = blockIdx.x * 256 + threadIdx.x;
  if (i >= BS * Dm) return;
  int bs = i >> 9, d = i & 511;
  float v = (d < CCc) ? feat[bs * CCc + d] : wemb[(long)word[bs] * WDd + (d - CCc)];
  x[i] = v;
  xb[i] = f2bf(v);
}

__global__ void k_hw_gate(float* __restrict__ x, u16b* __restrict__ xb,
                          const float* __restrict__ lin, const float* __restrict__ gat, int n) {
  int i = blockIdx.x * 256 + threadIdx.x;
  if (i >= n) return;
  float g = sigm(gat[i]);
  float h = fmaxf(lin[i], 0.f);
  float v = g * h + (1.f - g) * x[i];
  x[i] = v;
  xb[i] = f2bf(v);
}

// ---------------------------------------------------------------------------
// generic bf16 WMMA GEMM:  C[M,N] = A[M,K] * op(B) + bias0[n] + bias1[n]
//   BT=true : B stored [N,K] row-major (x @ W^T, all weight matmuls)
//   BT=false: B stored [K,N] row-major (a @ q)
// Double-buffered LDS tiles filled with GLOBAL_LOAD_ASYNC_TO_LDS_B128;
// edge tiles use clamped addresses (results masked at store), so EXEC stays
// all-ones and staging is branch-free. Batched via blockIdx.z.
// ---------------------------------------------------------------------------
template <bool BT>
__launch_bounds__(256)
__global__ void k_gemm(const u16b* __restrict__ A, long sA,
                       const u16b* __restrict__ Bw, long sB,
                       float* __restrict__ C, long sC,
                       int M, int N, int K,
                       const float* __restrict__ bias0, const float* __restrict__ bias1) {
  constexpr int BM = 64, BN = 128, BK = 32;
  constexpr int LDA  = BK + 8;   // 40 u16b (80B, 16B-aligned rows)
  constexpr int LDBT = BK + 8;   // 40   (NT: [BN][LDBT])
  constexpr int LDBN = BN + 8;   // 136  (NN: [BK][LDBN], 272B rows)
  __shared__ __align__(16) u16b As[2][BM * LDA];
  __shared__ __align__(16) u16b Bs[2][BT ? (BN * LDBT) : (BK * LDBN)];

  int tid = threadIdx.x;
  int wid = tid >> 5, lane = tid & 31, lr = lane & 15, hi = lane >> 4;
  int bM = blockIdx.y * BM, bN = blockIdx.x * BN;
  const u16b* Ab = A + (long)blockIdx.z * sA;
  const u16b* Bbp = Bw + (long)blockIdx.z * sB;
  float* Cb = C + (long)blockIdx.z * sC;

  int moff = (wid >> 2) * 32;   // 0 / 32
  int noff = (wid & 3) * 32;    // 0..96

  // per-thread staging coordinates (clamped to stay in-bounds)
  int arow = tid >> 2, ach = tid & 3;
  int agm = bM + arow; if (agm >= M) agm = M - 1;
  const u16b* aG = Ab + (long)agm * K + ach * 8;
  unsigned aL0 = (unsigned)(uintptr_t)&As[0][arow * LDA + ach * 8];
  unsigned aL1 = (unsigned)(uintptr_t)&As[1][arow * LDA + ach * 8];

  const u16b* bG[2];
  unsigned bL0[2], bL1[2];
  long bStep;  // global advance per BK step (in u16b elements)
  if constexpr (BT) {
#pragma unroll
    for (int rr = 0; rr < 2; rr++) {
      int row = (tid >> 2) + rr * 64, ch = tid & 3;
      int gn = bN + row; if (gn >= N) gn = N - 1;
      bG[rr] = Bbp + (long)gn * K + ch * 8;
      bL0[rr] = (unsigned)(uintptr_t)&Bs[0][row * LDBT + ch * 8];
      bL1[rr] = (unsigned)(uintptr_t)&Bs[1][row * LDBT + ch * 8];
    }
    bStep = BK;
  } else {
#pragma unroll
    for (int rr = 0; rr < 2; rr++) {
      int row = (tid >> 4) + rr * 16, ch = tid & 15;
      int gn = bN + ch * 8; if (gn + 8 > N) gn = N - 8;
      bG[rr] = Bbp + (long)row * N + gn;
      bL0[rr] = (unsigned)(uintptr_t)&Bs[0][row * LDBN + ch * 8];
      bL1[rr] = (unsigned)(uintptr_t)&Bs[1][row * LDBN + ch * 8];
    }
    bStep = (long)BK * N;
  }

  auto stage = [&](int it) {
    int buf = it & 1;
    long ka = (long)it * BK;
    async_ld_b128(buf ? aL1 : aL0, aG + ka);
#pragma unroll
    for (int rr = 0; rr < 2; rr++)
      async_ld_b128(buf ? bL1[rr] : bL0[rr], bG[rr] + (long)it * bStep);
  };

  v8f acc[2][2];
#pragma unroll
  for (int mi = 0; mi < 2; mi++)
#pragma unroll
    for (int ni = 0; ni < 2; ni++)
#pragma unroll
      for (int r = 0; r < 8; r++) acc[mi][ni][r] = 0.f;

  int nk = K / BK;
  stage(0);
  for (int it = 0; it < nk; it++) {
    if (it + 1 < nk) { stage(it + 1); wait_async3(); }
    else             { wait_async0(); }
    __syncthreads();   // all waves' current tiles are resident in LDS

    int cur = it & 1;
    Frag af[2], bfr[2];
#pragma unroll
    for (int mi = 0; mi < 2; mi++) {
      int m = moff + mi * 16 + lr;
      const u16b* p = &As[cur][m * LDA + hi * 16];
      uint4 t0 = *reinterpret_cast<const uint4*>(p);
      uint4 t1 = *reinterpret_cast<const uint4*>(p + 8);
      af[mi].u[0] = t0.x; af[mi].u[1] = t0.y; af[mi].u[2] = t0.z; af[mi].u[3] = t0.w;
      af[mi].u[4] = t1.x; af[mi].u[5] = t1.y; af[mi].u[6] = t1.z; af[mi].u[7] = t1.w;
    }
#pragma unroll
    for (int ni = 0; ni < 2; ni++) {
      int n = noff + ni * 16 + lr;
      if constexpr (BT) {
        const u16b* p = &Bs[cur][n * LDBT + hi * 16];
        uint4 t0 = *reinterpret_cast<const uint4*>(p);
        uint4 t1 = *reinterpret_cast<const uint4*>(p + 8);
        bfr[ni].u[0] = t0.x; bfr[ni].u[1] = t0.y; bfr[ni].u[2] = t0.z; bfr[ni].u[3] = t0.w;
        bfr[ni].u[4] = t1.x; bfr[ni].u[5] = t1.y; bfr[ni].u[6] = t1.z; bfr[ni].u[7] = t1.w;
      } else {
        int kb = hi * 16;
#pragma unroll
        for (int j = 0; j < 8; j++) {
          unsigned lo = Bs[cur][(kb + 2 * j) * LDBN + n];
          unsigned hv = Bs[cur][(kb + 2 * j + 1) * LDBN + n];
          bfr[ni].u[j] = lo | (hv << 16);
        }
      }
    }
#pragma unroll
    for (int mi = 0; mi < 2; mi++)
#pragma unroll
      for (int ni = 0; ni < 2; ni++)
        acc[mi][ni] = __builtin_amdgcn_wmma_f32_16x16x32_bf16(
            false, af[mi].v, false, bfr[ni].v, (short)0, acc[mi][ni], false, false);

    __syncthreads();   // tile consumed; buffer may be overwritten next round
  }

#pragma unroll
  for (int mi = 0; mi < 2; mi++)
#pragma unroll
    for (int ni = 0; ni < 2; ni++) {
      int nc = bN + noff + ni * 16 + lr;
      if (nc >= N) continue;
      float badd = 0.f;
      if (bias0) badd += bias0[nc];
      if (bias1) badd += bias1[nc];
#pragma unroll
      for (int r = 0; r < 8; r++) {
        int mr = bM + moff + mi * 16 + hi * 8 + r;
        if (mr < M) Cb[(long)mr * N + nc] = acc[mi][ni][r] + badd;
      }
    }
}

// ---------------------------------------------------------------------------
// persistent LSTM scan: one workgroup (1024 threads = 32 waves).
//   per step: z(32x1024) = h(32x256) @ whh^T via WMMA, then gates.
//   whh bf16 (1024x256) in global (L2-resident), h bf16 in LDS, c in regs.
// ---------------------------------------------------------------------------
constexpr int SCAN_LH = 264;   // h lds row stride (bf16)
constexpr int SCAN_LZ = 1032;  // z lds row stride (f32)
constexpr size_t SCAN_LDS = (size_t)32 * SCAN_LH * 2 + (size_t)32 * SCAN_LZ * 4;

template <bool REV>
__launch_bounds__(1024)
__global__ void k_lstm_scan(const u16b* __restrict__ whh,  // (1024,256) bf16
                            const float* __restrict__ xg,  // (B,T,1024)
                            float* __restrict__ out,       // (B,T,512)
                            int T, int outOff) {
  extern __shared__ __align__(16) char smem[];
  u16b* hL = (u16b*)smem;
  float* zL = (float*)(smem + (size_t)32 * SCAN_LH * 2);

  int tid = threadIdx.x;
  int wid = tid >> 5, lane = tid & 31, lr = lane & 15, hi = lane >> 4;
  for (int i = tid; i < 32 * SCAN_LH; i += 1024) hL[i] = 0;
  float cst[8];
#pragma unroll
  for (int j = 0; j < 8; j++) cst[j] = 0.f;
  int nn = tid & 255, bg = tid >> 8;
  __syncthreads();

  for (int s = 0; s < T; s++) {
    int t = REV ? (T - 1 - s) : s;
    // --- GEMM phase: each wave computes 4 of the 128 (16x16) tiles
#pragma unroll
    for (int q = 0; q < 4; q++) {
      int tile = wid * 4 + q;
      int mi = tile & 1;     // 0..1
      int ni = tile >> 1;    // 0..63
      v8f acc;
#pragma unroll
      for (int r = 0; r < 8; r++) acc[r] = 0.f;
#pragma unroll
      for (int kk = 0; kk < 8; kk++) {
        Frag a, b;
        {
          int m = mi * 16 + lr;
          const u16b* p = &hL[m * SCAN_LH + kk * 32 + hi * 16];
          uint4 t0 = *reinterpret_cast<const uint4*>(p);
          uint4 t1 = *reinterpret_cast<const uint4*>(p + 8);
          a.u[0] = t0.x; a.u[1] = t0.y; a.u[2] = t0.z; a.u[3] = t0.w;
          a.u[4] = t1.x; a.u[5] = t1.y; a.u[6] = t1.z; a.u[7] = t1.w;
        }
        {
          int n = ni * 16 + lr;
          const u16b* p = whh + (long)n * Hh + kk * 32 + hi * 16;
          uint4 t0 = *reinterpret_cast<const uint4*>(p);
          uint4 t1 = *reinterpret_cast<const uint4*>(p + 8);
          b.u[0] = t0.x; b.u[1] = t0.y; b.u[2] = t0.z; b.u[3] = t0.w;
          b.u[4] = t1.x; b.u[5] = t1.y; b.u[6] = t1.z; b.u[7] = t1.w;
        }
        acc = __builtin_amdgcn_wmma_f32_16x16x32_bf16(
            false, a.v, false, b.v, (short)0, acc, false, false);
      }
#pragma unroll
      for (int r = 0; r < 8; r++) {
        int m = mi * 16 + hi * 8 + r;
        int n = ni * 16 + lr;
        zL[m * SCAN_LZ + n] = acc[r];
      }
    }
    __syncthreads();
    // --- gate phase: thread owns (b = bg*8+j, n = nn)
#pragma unroll
    for (int j = 0; j < 8; j++) {
      int b = bg * 8 + j;
      long xo = ((long)b * T + t) * 1024 + nn;
      float zi = zL[b * SCAN_LZ + nn]       + xg[xo];
      float zf = zL[b * SCAN_LZ + nn + 256] + xg[xo + 256];
      float zg = zL[b * SCAN_LZ + nn + 512] + xg[xo + 512];
      float zo = zL[b * SCAN_LZ + nn + 768] + xg[xo + 768];
      float c = sigm(zf) * cst[j] + sigm(zi) * tanhf(zg);
      cst[j] = c;
      float h = sigm(zo) * tanhf(c);
      out[((long)b * T + t) * Dm + outOff + nn] = h;
      hL[b * SCAN_LH + nn] = f2bf(h);
    }
    __syncthreads();
  }
}

// ---------------------------------------------------------------------------
// attention / reduction kernels
// ---------------------------------------------------------------------------
__global__ void k_rowdot(const float* __restrict__ X, const float* __restrict__ w,
                         const float* __restrict__ bptr, float* __restrict__ outv, int Dlen) {
  __shared__ float red[256];
  int row = blockIdx.x, t = threadIdx.x;
  float s = 0.f;
  for (int d = t; d < Dlen; d += 256) s += X[(long)row * Dlen + d] * w[d];
  red[t] = s;
  __syncthreads();
  for (int k = 128; k > 0; k >>= 1) {
    if (t < k) red[t] += red[t + k];
    __syncthreads();
  }
  if (!t) outv[row] = red[0] + bptr[0];
}

__global__ void k_s_add(float* __restrict__ s, const float* __restrict__ cw,
                        const float* __restrict__ qw, const float* __restrict__ cqb) {
  int i = blockIdx.x * 256 + threadIdx.x;
  if (i >= Bb * CLl * QLl) return;
  int j = i & 63;
  int bi = i >> 6;           // b*CL + ic
  int b = bi / CLl;
  s[i] += cw[bi] + qw[b * QLl + j] + cqb[0];
}

__global__ void k_softmax_j(const float* __restrict__ s, u16b* __restrict__ a,
                            float* __restrict__ rmax) {
  __shared__ float red[64];
  int bi = blockIdx.x;
  int j = threadIdx.x;  // 64 threads
  float v = s[(long)bi * 64 + j];
  red[j] = v;
  __syncthreads();
  for (int k = 32; k > 0; k >>= 1) {
    if (j < k) red[j] = fmaxf(red[j], red[j + k]);
    __syncthreads();
  }
  float mx = red[0];
  __syncthreads();
  float e = __expf(v - mx);
  red[j] = e;
  __syncthreads();
  for (int k = 32; k > 0; k >>= 1) {
    if (j < k) red[j] += red[j + k];
    __syncthreads();
  }
  a[(long)bi * 64 + j] = f2bf(e / red[0]);
  if (!j) rmax[bi] = mx;
}

__global__ void k_softmax_i(const float* __restrict__ rmax, float* __restrict__ bbv) {
  __shared__ float red[512];
  int b = blockIdx.x, i = threadIdx.x;  // 512 threads
  float v = (i < CLl) ? rmax[b * CLl + i] : -3.4e38f;
  red[i] = v;
  __syncthreads();
  for (int k = 256; k > 0; k >>= 1) {
    if (i < k) red[i] = fmaxf(red[i], red[i + k]);
    __syncthreads();
  }
  float mx = red[0];
  __syncthreads();
  float e = (i < CLl) ? __expf(v - mx) : 0.f;
  red[i] = e;
  __syncthreads();
  for (int k = 256; k > 0; k >>= 1) {
    if (i < k) red[i] += red[i + k];
    __syncthreads();
  }
  if (i < CLl) bbv[b * CLl + i] = e / red[0];
}

__global__ void k_q2c(const float* __restrict__ bbv, const float* __restrict__ c,
                      float* __restrict__ q2c) {
  int b = blockIdx.x;
  int d = blockIdx.y * 256 + threadIdx.x;  // 0..511
  float s = 0.f;
  for (int i = 0; i < CLl; i++) s += bbv[b * CLl + i] * c[((long)b * CLl + i) * Dm + d];
  q2c[b * Dm + d] = s;
}

__global__ void k_build_g(const float* __restrict__ c, const float* __restrict__ c2q,
                          const float* __restrict__ q2c, float* __restrict__ g,
                          u16b* __restrict__ gb) {
  long i = (long)blockIdx.x * 256 + threadIdx.x;
  if (i >= (long)Bb * CLl * Dm) return;
  long bs = i >> 9;
  int d = (int)(i & 511);
  int b = (int)(bs / CLl);
  float cv = c[i], c2 = c2q[i], qc = q2c[b * Dm + d];
  long base = bs * DD + d;
  float v0 = cv, v1 = c2, v2 = cv * c2, v3 = cv * qc;
  g[base] = v0; g[base + 512] = v1; g[base + 1024] = v2; g[base + 1536] = v3;
  gb[base] = f2bf(v0); gb[base + 512] = f2bf(v1);
  gb[base + 1024] = f2bf(v2); gb[base + 1536] = f2bf(v3);
}

__global__ void k_pout(const float* __restrict__ g, const float* __restrict__ gw,
                       const float* __restrict__ gbias, const float* __restrict__ m,
                       const float* __restrict__ mw, const float* __restrict__ mbias,
                       float* __restrict__ p) {
  __shared__ float red[256];
  int bs = blockIdx.x, t = threadIdx.x;
  float s = 0.f;
  for (int d = t; d < DD; d += 256) s += g[(long)bs * DD + d] * gw[d];
  for (int d = t; d < Dm; d += 256) s += m[(long)bs * Dm + d] * mw[d];
  red[t] = s;
  __syncthreads();
  for (int k = 128; k > 0; k >>= 1) {
    if (t < k) red[t] += red[t + k];
    __syncthreads();
  }
  if (!t) p[bs] = red[0] + gbias[0] + mbias[0];
}

// ---------------------------------------------------------------------------
// host orchestration
// ---------------------------------------------------------------------------
enum {
  IN_CHAR_EMB = 0, IN_CONV_W, IN_CONV_B, IN_WORD_EMB,
  IN_HW0_LW, IN_HW0_LB, IN_HW0_GW, IN_HW0_GB,
  IN_HW1_LW, IN_HW1_LB, IN_HW1_GW, IN_HW1_GB,
  IN_CTX = 12,   // wih, whh, bih, bhh, wih_r, whh_r, bih_r, bhh_r
  IN_MOD1 = 20,
  IN_MOD2 = 28,
  IN_OUT = 36,
  IN_ATT_C_W = 44, IN_ATT_C_B, IN_ATT_Q_W, IN_ATT_Q_B, IN_ATT_CQ_W, IN_ATT_CQ_B,
  IN_P1G_W = 50, IN_P1G_B, IN_P1M_W, IN_P1M_B,
  IN_P2G_W = 54, IN_P2G_B, IN_P2M_W, IN_P2M_B,
  IN_C_CHAR = 58, IN_Q_CHAR, IN_C_WORD, IN_Q_WORD
};

extern "C" void kernel_launch(void* const* d_in, const int* in_sizes, int n_in,
                              void* d_out, int out_size, void* d_ws, size_t ws_size,
                              hipStream_t stream) {
  (void)in_sizes; (void)n_in; (void)out_size; (void)ws_size;
  auto F = [&](int i) { return (const float*)d_in[i]; };
  auto I = [&](int i) { return (const int*)d_in[i]; };

  char* wsbase = (char*)d_ws;
  size_t off = 0;
  auto alloc = [&](size_t bytes) -> char* {
    size_t a = (off + 255) & ~(size_t)255;
    char* p = wsbase + a;
    off = a + bytes;
    return p;
  };
  auto fal = [&](long n) { return (float*)alloc((size_t)n * 4); };
  auto bal = [&](long n) { return (u16b*)alloc((size_t)n * 2); };

  const long BSc = (long)Bb * CLl;  // 12800
  const long BSq = (long)Bb * QLl;  // 2048

  // f32 scratch
  float* featc = fal(BSc * CCc);
  float* featq = fal(BSq * CCc);
  float* xc   = fal(BSc * Dm);
  float* xq   = fal(BSq * Dm);
  float* lin  = fal(BSc * Dm);
  float* gat  = fal(BSc * Dm);
  float* xgF  = fal(BSc * 1024);
  float* xgR  = fal(BSc * 1024);
  float* xgqF = fal(BSq * 1024);
  float* xgqR = fal(BSq * 1024);
  float* cctx = fal(BSc * Dm);
  float* qctx = fal(BSq * Dm);
  float* sbuf = fal(BSc * QLl);
  float* cwv  = fal(BSc);
  float* qwv  = fal(BSq);
  float* rmax = fal(BSc);
  float* bbv  = fal(BSc);
  float* c2q  = fal(BSc * Dm);
  float* q2cb = fal((long)Bb * Dm);
  float* gmat = fal(BSc * DD);
  float* m1   = fal(BSc * Dm);
  float* mm   = fal(BSc * Dm);
  float* m2   = fal(BSc * Dm);

  // bf16 scratch
  u16b* bxc = bal(BSc * Dm);
  u16b* bxq = bal(BSq * Dm);
  u16b* bcw = bal(BSc * Dm);
  u16b* bq  = bal(BSq * Dm);
  u16b* ba  = bal(BSc * QLl);
  u16b* bg  = bal(BSc * DD);
  u16b* bm  = bal(BSc * Dm);

  // bf16 weights
  u16b* w_lw0 = bal(512 * 512); u16b* w_gw0 = bal(512 * 512);
  u16b* w_lw1 = bal(512 * 512); u16b* w_gw1 = bal(512 * 512);
  u16b* w_ctx_wih  = bal(1024 * 512);  u16b* w_ctx_whh  = bal(1024 * 256);
  u16b* w_ctx_wihr = bal(1024 * 512);  u16b* w_ctx_whhr = bal(1024 * 256);
  u16b* w_m1_wih   = bal(1024 * 2048); u16b* w_m1_whh   = bal(1024 * 256);
  u16b* w_m1_wihr  = bal(1024 * 2048); u16b* w_m1_whhr  = bal(1024 * 256);
  u16b* w_m2_wih   = bal(1024 * 512);  u16b* w_m2_whh   = bal(1024 * 256);
  u16b* w_m2_wihr  = bal(1024 * 512);  u16b* w_m2_whhr  = bal(1024 * 256);
  u16b* w_o_wih    = bal(1024 * 512);  u16b* w_o_whh    = bal(1024 * 256);
  u16b* w_o_wihr   = bal(1024 * 512);  u16b* w_o_whhr   = bal(1024 * 256);

  auto cvt = [&](const float* src, u16b* dst, long n) {
    k_f32_to_bf16<<<(int)((n + 255) / 256), 256, 0, stream>>>(src, dst, (int)n);
  };
  auto gemmNT = [&](const u16b* A, long sA, const u16b* W, long sW, float* C, long sC,
                    int M, int N, int K, int batch, const float* b0, const float* b1) {
    dim3 grid((N + 127) / 128, (M + 63) / 64, batch);
    k_gemm<true><<<grid, 256, 0, stream>>>(A, sA, W, sW, C, sC, M, N, K, b0, b1);
  };
  auto gemmNN = [&](const u16b* A, long sA, const u16b* W, long sW, float* C, long sC,
                    int M, int N, int K, int batch) {
    dim3 grid((N + 127) / 128, (M + 63) / 64, batch);
    k_gemm<false><<<grid, 256, 0, stream>>>(A, sA, W, sW, C, sC, M, N, K,
                                            (const float*)nullptr, (const float*)nullptr);
  };
  auto bilstm = [&](int base, int Kin, const u16b* Abf, int T,
                    const u16b* wih, const u16b* whh, const u16b* wihr, const u16b* whhr,
                    float* xgf, float* xgr, float* outp) {
    gemmNT(Abf, 0, wih,  0, xgf, 0, Bb * T, 1024, Kin, 1, F(base + 2), F(base + 3));
    gemmNT(Abf, 0, wihr, 0, xgr, 0, Bb * T, 1024, Kin, 1, F(base + 6), F(base + 7));
    k_lstm_scan<false><<<1, 1024, SCAN_LDS, stream>>>(whh,  xgf, outp, T, 0);
    k_lstm_scan<true ><<<1, 1024, SCAN_LDS, stream>>>(whhr, xgr, outp, T, Hh);
  };

  // ---- 1. weight conversion to bf16 ----
  cvt(F(IN_HW0_LW), w_lw0, 512 * 512); cvt(F(IN_HW0_GW), w_gw0, 512 * 512);
  cvt(F(IN_HW1_LW), w_lw1, 512 * 512); cvt(F(IN_HW1_GW), w_gw1, 512 * 512);
  cvt(F(IN_CTX + 0), w_ctx_wih,  1024 * 512);  cvt(F(IN_CTX + 1), w_ctx_whh,  1024 * 256);
  cvt(F(IN_CTX + 4), w_ctx_wihr, 1024 * 512);  cvt(F(IN_CTX + 5), w_ctx_whhr, 1024 * 256);
  cvt(F(IN_MOD1 + 0), w_m1_wih,  1024 * 2048); cvt(F(IN_MOD1 + 1), w_m1_whh,  1024 * 256);
  cvt(F(IN_MOD1 + 4), w_m1_wihr, 1024 * 2048); cvt(F(IN_MOD1 + 5), w_m1_whhr, 1024 * 256);
  cvt(F(IN_MOD2 + 0), w_m2_wih,  1024 * 512);  cvt(F(IN_MOD2 + 1), w_m2_whh,  1024 * 256);
  cvt(F(IN_MOD2 + 4), w_m2_wihr, 1024 * 512);  cvt(F(IN_MOD2 + 5), w_m2_whhr, 1024 * 256);
  cvt(F(IN_OUT + 0), w_o_wih,  1024 * 512);    cvt(F(IN_OUT + 1), w_o_whh,  1024 * 256);
  cvt(F(IN_OUT + 4), w_o_wihr, 1024 * 512);    cvt(F(IN_OUT + 5), w_o_whhr, 1024 * 256);

  // ---- 2. char conv + concat ----
  k_char_conv<<<(int)BSc, 256, 0, stream>>>(I(IN_C_CHAR), F(IN_CHAR_EMB), F(IN_CONV_W),
                                            F(IN_CONV_B), featc);
  k_char_conv<<<(int)BSq, 256, 0, stream>>>(I(IN_Q_CHAR), F(IN_CHAR_EMB), F(IN_CONV_W),
                                            F(IN_CONV_B), featq);
  k_concat<<<(int)((BSc * Dm + 255) / 256), 256, 0, stream>>>(featc, I(IN_C_WORD),
                                                              F(IN_WORD_EMB), xc, bxc, (int)BSc);
  k_concat<<<(int)((BSq * Dm + 255) / 256), 256, 0, stream>>>(featq, I(IN_Q_WORD),
                                                              F(IN_WORD_EMB), xq, bxq, (int)BSq);

  // ---- 3. highway x2 (c and q) ----
  for (int l = 0; l < 2; l++) {
    int bidx = l ? IN_HW1_LW : IN_HW0_LW;
    u16b* lw = l ? w_lw1 : w_lw0;
    u16b* gw = l ? w_gw1 : w_gw0;
    gemmNT(bxc, 0, lw, 0, lin, 0, (int)BSc, 512, 512, 1, F(bidx + 1), nullptr);
    gemmNT(bxc, 0, gw, 0, gat, 0, (int)BSc, 512, 512, 1, F(bidx + 3), nullptr);
    k_hw_gate<<<(int)((BSc * Dm + 255) / 256), 256, 0, stream>>>(xc, bxc, lin, gat,
                                                                 (int)(BSc * Dm));
    gemmNT(bxq, 0, lw, 0, lin, 0, (int)BSq, 512, 512, 1, F(bidx + 1), nullptr);
    gemmNT(bxq, 0, gw, 0, gat, 0, (int)BSq, 512, 512, 1, F(bidx + 3), nullptr);
    k_hw_gate<<<(int)((BSq * Dm + 255) / 256), 256, 0, stream>>>(xq, bxq, lin, gat,
                                                                 (int)(BSq * Dm));
  }

  // ---- 4. context BiLSTM ----
  bilstm(IN_CTX, 512, bxc, CLl, w_ctx_wih, w_ctx_whh, w_ctx_wihr, w_ctx_whhr, xgF, xgR, cctx);
  bilstm(IN_CTX, 512, bxq, QLl, w_ctx_wih, w_ctx_whh, w_ctx_wihr, w_ctx_whhr, xgqF, xgqR, qctx);

  // ---- 5. attention ----
  k_rowdot<<<(int)BSc, 256, 0, stream>>>(cctx, F(IN_ATT_C_W), F(IN_ATT_C_B), cwv, Dm);
  k_rowdot<<<(int)BSq, 256, 0, stream>>>(qctx, F(IN_ATT_Q_W), F(IN_ATT_Q_B), qwv, Dm);
  k_scale_bf16<<<(int)((BSc * Dm + 255) / 256), 256, 0, stream>>>(cctx, F(IN_ATT_CQ_W), bcw,
                                                                  (int)(BSc * Dm), Dm - 1);
  cvt(qctx, bq, BSq * Dm);
  gemmNT(bcw, (long)CLl * Dm, bq, (long)QLl * Dm, sbuf, (long)CLl * QLl,
         CLl, QLl, Dm, Bb, nullptr, nullptr);
  k_s_add<<<(int)((BSc * QLl + 255) / 256), 256, 0, stream>>>(sbuf, cwv, qwv, F(IN_ATT_CQ_B));
  k_softmax_j<<<(int)BSc, 64, 0, stream>>>(sbuf, ba, rmax);
  k_softmax_i<<<Bb, 512, 0, stream>>>(rmax, bbv);
  gemmNN(ba, (long)CLl * QLl, bq, (long)QLl * Dm, c2q, (long)CLl * Dm, CLl, Dm, QLl, Bb);
  k_q2c<<<dim3(Bb, 2), 256, 0, stream>>>(bbv, cctx, q2cb);
  k_build_g<<<(int)((BSc * Dm + 255) / 256), 256, 0, stream>>>(cctx, c2q, q2cb, gmat, bg);

  // ---- 6. modeling LSTMs ----
  bilstm(IN_MOD1, 2048, bg, CLl, w_m1_wih, w_m1_whh, w_m1_wihr, w_m1_whhr, xgF, xgR, m1);
  cvt(m1, bm, BSc * Dm);
  bilstm(IN_MOD2, 512, bm, CLl, w_m2_wih, w_m2_whh, w_m2_wihr, w_m2_whhr, xgF, xgR, mm);

  // ---- 7. p1 ----
  k_pout<<<(int)BSc, 256, 0, stream>>>(gmat, F(IN_P1G_W), F(IN_P1G_B), mm,
                                       F(IN_P1M_W), F(IN_P1M_B), (float*)d_out);

  // ---- 8. output LSTM + p2 ----
  cvt(mm, bm, BSc * Dm);
  bilstm(IN_OUT, 512, bm, CLl, w_o_wih, w_o_whh, w_o_wihr, w_o_whhr, xgF, xgR, m2);
  k_pout<<<(int)BSc, 256, 0, stream>>>(gmat, F(IN_P2G_W), F(IN_P2G_B), m2,
                                       F(IN_P2M_W), F(IN_P2M_B), (float*)d_out + BSc);
}
